// SurrogateGCN_85985245266460
// MI455X (gfx1250) — matrix operand order repaired and checked by reading, not compiled, not run on hardware
//
#include <hip/hip_runtime.h>
#include <hip/hip_bf16.h>

// ---------------------------------------------------------------------------
// GCN surrogate for MI455X (gfx1250, wave32, WMMA).
//   h1 = agg(x @ W1) + b1 ; h2 = agg(h1 @ W2) + b2 ; out = log_softmax(h2 @ Wlin + blin)
// GEMMs run through v_wmma_f32_16x16x32_f16 (f32 accumulate).
// ---------------------------------------------------------------------------

typedef __attribute__((ext_vector_type(16))) _Float16 v16h;
typedef __attribute__((ext_vector_type(8)))  _Float16 v8h;
typedef __attribute__((ext_vector_type(8)))  float    v8f;

#define F_DIM 256   // F_IN == F_HID == 256
#define O_DIM 40    // F_OUT

// ---- degree / normalization -----------------------------------------------
__global__ void deg_init_kernel(float* deg, int n) {
    int i = blockIdx.x * blockDim.x + threadIdx.x;
    if (i < n) deg[i] = 1.0f;                 // self-loop contributes 1
}

__global__ void deg_accum_kernel(const long long* __restrict__ dst, float* deg, int e) {
    int i = blockIdx.x * blockDim.x + threadIdx.x;
    if (i < e) atomicAdd(&deg[dst[i]], 1.0f);
}

__global__ void deg_rsqrt_kernel(float* deg, int n) {
    int i = blockIdx.x * blockDim.x + threadIdx.x;
    if (i < n) deg[i] = rsqrtf(deg[i]);       // deg >= 1 always (self-loops)
}

// ---- f32 -> f16 conversion -------------------------------------------------
__global__ void cvt_f16_kernel(const float* __restrict__ in, _Float16* __restrict__ out, int n) {
    int i = blockIdx.x * blockDim.x + threadIdx.x;
    if (i < n) out[i] = (_Float16)in[i];
}

// ---- pack W (K x N, row-major f32) into WMMA B fragments (f16) -------------
// Fragment layout [K/32][N/16][lane 0..31][j 0..15]:
//   lanes 0..15  : N = nt*16 + lane,      K = kt*32 + (j&7) + (j>>3)*16      (j = 0..15)
//   lanes 16..31 : N = nt*16 + lane - 16, K = kt*32 + 8 + (j&7) + (j>>3)*16
__global__ void pack_b_kernel(const float* __restrict__ W, _Float16* __restrict__ Bp,
                              int K, int N) {
    int idx = blockIdx.x * blockDim.x + threadIdx.x;
    if (idx >= K * N) return;
    int j    = idx & 15;
    int l    = (idx >> 4) & 31;
    int frag = idx >> 9;
    int ntn  = N >> 4;
    int kt = frag / ntn, nt = frag % ntn;
    int n = nt * 16 + (l & 15);
    int k = kt * 32 + (l >> 4) * 8 + (j & 7) + (j >> 3) * 16;
    Bp[idx] = (_Float16)W[k * N + n];
}

// ---- WMMA GEMM: C[M x N] = A[M x K] (f16 row-major) * Bp (packed f16) ------
// One wave per 16x16 C tile; K-loop in steps of 32 -> 8 v_wmma per tile.
__global__ void gemm_wmma_kernel(const _Float16* __restrict__ A,
                                 const _Float16* __restrict__ Bp,
                                 float* __restrict__ C,
                                 int M, int N, int K) {
    int gtid  = blockIdx.x * blockDim.x + threadIdx.x;
    int wave  = gtid >> 5;
    int lane  = threadIdx.x & 31;
    int ntn   = N >> 4;
    int total = (M >> 4) * ntn;
    if (wave >= total) return;                // wave-uniform: EXEC stays all-ones
    int tm = wave / ntn;
    int tn = wave % ntn;

    // A fragment addressing (ISA 16-bit A 16x32 layout):
    //   lanes 0..15 : M = tm*16+lane,    K chunks [k0+0..7]  and [k0+16..23]
    //   lanes 16..31: M = tm*16+lane-16, K chunks [k0+8..15] and [k0+24..31]
    int row   = tm * 16 + (lane & 15);
    int khalf = (lane >> 4) * 8;
    const _Float16* arow = A + (long long)row * K + khalf;

    v8f acc = {};
    for (int k0 = 0; k0 < K; k0 += 32) {
        const _Float16* ap = arow + k0;
        __builtin_prefetch(ap + 32, 0, 1);    // global_prefetch_b8 for next K step
        v8h a0 = *(const v8h*)(ap);
        v8h a1 = *(const v8h*)(ap + 16);
        v16h a = __builtin_shufflevector(a0, a1,
                                         0, 1, 2, 3, 4, 5, 6, 7,
                                         8, 9, 10, 11, 12, 13, 14, 15);
        const _Float16* bp = Bp + ((( (k0 >> 5) * ntn + tn) * 32 + lane) << 4);
        v16h b = *(const v16h*)bp;
        acc = __builtin_amdgcn_wmma_f32_16x16x32_f16(
                  /*neg_a=*/false, a, /*neg_b=*/false, b,
                  /*c_mod=*/(short)0, acc, /*reuse_a=*/false, /*reuse_b=*/false);
    }

    // C/D layout: VGPR r -> lanes 0..15: M=r, N=lane; lanes 16..31: M=r+8, N=lane-16
    int col   = tn * 16 + (lane & 15);
    int rbase = tm * 16 + ((lane >> 4) ? 8 : 0);
#pragma unroll
    for (int r = 0; r < 8; ++r)
        C[(long long)(rbase + r) * N + col] = acc[r];
}

// ---- aggregation init: agg[i,f] = bias[f] + dinv[i]^2 * h[i,f] (self-loop) -
__global__ void agg_init_kernel(const float* __restrict__ h,
                                const float* __restrict__ dinv,
                                const float* __restrict__ bias,
                                float* __restrict__ agg, int nf) {
    int idx = blockIdx.x * blockDim.x + threadIdx.x;
    if (idx >= nf) return;
    int i = idx >> 8;          // / F_DIM
    int f = idx & 255;         // % F_DIM
    float d = dinv[i];
    agg[idx] = bias[f] + d * d * h[idx];
}

// ---- edge scatter: agg[dst] += h[src] * dinv[src]*dinv[dst] ----------------
// One wave per edge; each lane handles 8 contiguous floats (32*8 = 256).
__global__ void edge_scatter_kernel(const long long* __restrict__ src,
                                    const long long* __restrict__ dst,
                                    const float* __restrict__ h,
                                    const float* __restrict__ dinv,
                                    float* __restrict__ agg, int e) {
    int gtid = blockIdx.x * blockDim.x + threadIdx.x;
    int edge = gtid >> 5;
    if (edge >= e) return;
    int lane = gtid & 31;
    long long s = src[edge];
    long long d = dst[edge];
    float c = dinv[s] * dinv[d];
    const float4* hp = (const float4*)(h + s * F_DIM) + lane * 2;
    float* ap = agg + d * F_DIM + lane * 8;
    float4 v0 = hp[0];
    float4 v1 = hp[1];
    atomicAdd(ap + 0, v0.x * c);
    atomicAdd(ap + 1, v0.y * c);
    atomicAdd(ap + 2, v0.z * c);
    atomicAdd(ap + 3, v0.w * c);
    atomicAdd(ap + 4, v1.x * c);
    atomicAdd(ap + 5, v1.y * c);
    atomicAdd(ap + 6, v1.z * c);
    atomicAdd(ap + 7, v1.w * c);
}

// ---- head: logits = h @ Wlin + blin ; out = log_softmax(logits) ------------
// One block (64 threads) per node.
__global__ void head_kernel(const float* __restrict__ h,
                            const float* __restrict__ Wlin,
                            const float* __restrict__ blin,
                            float* __restrict__ out, int n) {
    __shared__ float sh[F_DIM];
    __shared__ float slog[64];
    __shared__ float red[2];
    int node = blockIdx.x;
    if (node >= n) return;
    int t = threadIdx.x;
    for (int i = t; i < F_DIM; i += 64) sh[i] = h[(long long)node * F_DIM + i];
    __syncthreads();
    if (t < O_DIM) {
        float v = blin[t];
        for (int k = 0; k < F_DIM; ++k) v += sh[k] * Wlin[k * O_DIM + t];
        slog[t] = v;
    }
    __syncthreads();
    if (t == 0) {
        float m = slog[0];
        for (int i = 1; i < O_DIM; ++i) m = fmaxf(m, slog[i]);
        float s = 0.0f;
        for (int i = 0; i < O_DIM; ++i) s += __expf(slog[i] - m);
        red[0] = m;
        red[1] = __logf(s);
    }
    __syncthreads();
    if (t < O_DIM) out[(long long)node * O_DIM + t] = slog[t] - red[0] - red[1];
}

// ---------------------------------------------------------------------------
extern "C" void kernel_launch(void* const* d_in, const int* in_sizes, int n_in,
                              void* d_out, int out_size, void* d_ws, size_t ws_size,
                              hipStream_t stream) {
    const float*      x    = (const float*)d_in[0];
    const long long*  eidx = (const long long*)d_in[1];
    const float*      W1   = (const float*)d_in[2];
    const float*      b1   = (const float*)d_in[3];
    const float*      W2   = (const float*)d_in[4];
    const float*      b2   = (const float*)d_in[5];
    const float*      Wlin = (const float*)d_in[6];
    const float*      blin = (const float*)d_in[7];
    float*            out  = (float*)d_out;

    const int N = in_sizes[0] / F_DIM;        // 100000
    const int E = in_sizes[1] / 2;            // 3200000
    const int NF = N * F_DIM;

    const long long* src = eidx;
    const long long* dst = eidx + E;

    // Workspace carve-out (256B-aligned blocks)
    char* base = (char*)d_ws;
    size_t p = 0;
    auto carve = [&](size_t bytes) -> char* {
        char* q = base + p;
        p = (p + bytes + 255) & ~(size_t)255;
        return q;
    };
    float*    dinv = (float*)   carve((size_t)N * 4);             // deg then rsqrt
    _Float16* xh   = (_Float16*)carve((size_t)NF * 2);            // f16 activations
    _Float16* w1p  = (_Float16*)carve((size_t)F_DIM * F_DIM * 2); // packed W1
    _Float16* w2p  = (_Float16*)carve((size_t)F_DIM * F_DIM * 2); // packed W2
    float*    hbuf = (float*)   carve((size_t)NF * 4);            // GEMM out
    float*    agg  = (float*)   carve((size_t)NF * 4);            // aggregated

    const int TB = 256;
    int gN   = (N + TB - 1) / TB;
    int gE   = (E + TB - 1) / TB;
    int gNF  = (NF + TB - 1) / TB;
    int gW   = (F_DIM * F_DIM + TB - 1) / TB;
    int waves = (N / 16) * (F_DIM / 16);
    int gGemm = (waves * 32 + TB - 1) / TB;
    int gScat = (E * 32 + TB - 1) / TB;       // E*32 = 102.4M threads

    // 1) degree / normalization
    deg_init_kernel<<<gN, TB, 0, stream>>>(dinv, N);
    deg_accum_kernel<<<gE, TB, 0, stream>>>(dst, dinv, E);
    deg_rsqrt_kernel<<<gN, TB, 0, stream>>>(dinv, N);

    // 2) precision staging
    cvt_f16_kernel<<<gNF, TB, 0, stream>>>(x, xh, NF);
    pack_b_kernel<<<gW, TB, 0, stream>>>(W1, w1p, F_DIM, F_DIM);
    pack_b_kernel<<<gW, TB, 0, stream>>>(W2, w2p, F_DIM, F_DIM);

    // 3) conv1: GEMM + normalized aggregation + bias
    gemm_wmma_kernel<<<gGemm, TB, 0, stream>>>(xh, w1p, hbuf, N, F_DIM, F_DIM);
    agg_init_kernel<<<gNF, TB, 0, stream>>>(hbuf, dinv, b1, agg, NF);
    edge_scatter_kernel<<<gScat, TB, 0, stream>>>(src, dst, hbuf, dinv, agg, E);

    // 4) conv2
    cvt_f16_kernel<<<gNF, TB, 0, stream>>>(agg, xh, NF);
    gemm_wmma_kernel<<<gGemm, TB, 0, stream>>>(xh, w2p, hbuf, N, F_DIM, F_DIM);
    agg_init_kernel<<<gNF, TB, 0, stream>>>(hbuf, dinv, b2, agg, NF);
    edge_scatter_kernel<<<gScat, TB, 0, stream>>>(src, dst, hbuf, dinv, agg, E);

    // 5) head: linear + log_softmax
    head_kernel<<<N, 64, 0, stream>>>(agg, Wlin, blin, out, N);
}
